// EdgeGCN_LSTM_8650064134828
// MI455X (gfx1250) — matrix operand order repaired and checked by reading, not compiled, not run on hardware
//
#include <hip/hip_runtime.h>
#include <hip/hip_bf16.h>

typedef __attribute__((ext_vector_type(16))) _Float16 v16h;
typedef __attribute__((ext_vector_type(8)))  _Float16 v8h;
typedef __attribute__((ext_vector_type(8)))  float    v8f;

#define NN   10000
#define EE   100000
#define HH   128
#define LHH  256
#define G4   1024      // 4*LH
#define KEF  288       // 258 padded to multiple of 32
#define BNEPS 1e-5f

// ---------------------------------------------------------------- utilities
__global__ void zero_f32(float* p, int n) {
  int i = blockIdx.x * blockDim.x + threadIdx.x;
  if (i < n) p[i] = 0.0f;
}

__global__ void add2_f32(const float* a, const float* b, float* d, int n) {
  int i = blockIdx.x * blockDim.x + threadIdx.x;
  if (i < n) d[i] = a[i] + b[i];
}

__global__ void cvt_f32_f16(const float* src, _Float16* dst, int n) {
  int i = blockIdx.x * blockDim.x + threadIdx.x;
  if (i < n) dst[i] = (_Float16)src[i];
}

// whhT[k*1024 + j] = whh[j*256 + k]
__global__ void transp_whh(const float* whh, float* whhT) {
  int i = blockIdx.x * blockDim.x + threadIdx.x;
  if (i < 256 * G4) {
    int k = i & 255, j = i >> 8;
    whhT[k * G4 + j] = whh[j * 256 + k];
  }
}

// --------------------------------------------------------------- GCN stage
__global__ void deg_edges(const int* col, float* deg, int E) {
  int e = blockIdx.x * blockDim.x + threadIdx.x;
  if (e < E) atomicAdd(&deg[col[e]], 1.0f);
}

__global__ void dinv_fin(float* deg, int N) {   // in-place: count(+self) -> deg^-1/2
  int i = blockIdx.x * blockDim.x + threadIdx.x;
  if (i < N) {
    float d = deg[i] + 1.0f;                    // self loop
    deg[i] = (d > 0.0f) ? rsqrtf(d) : 0.0f;
  }
}

// layer-1 is rank-1: sacc[c] = sum_e norm*x[r]  (+ self term)
__global__ void gcn1_scatter(const int* row, const int* col, const float* x,
                             const float* dinv, float* sacc, int E, int N) {
  int idx = blockIdx.x * blockDim.x + threadIdx.x;
  if (idx < E) {
    int r = row[idx], c = col[idx];
    atomicAdd(&sacc[c], dinv[r] * dinv[c] * x[r]);
  } else if (idx < E + N) {
    int i = idx - E;
    atomicAdd(&sacc[i], dinv[i] * dinv[i] * x[i]);
  }
}

// h1[n][f] = relu(BN(sacc[n]*W1[f] + b1[f]))
__global__ void h1_fused(const float* sacc, const float* W1, const float* b1,
                         const float* g1, const float* be1, const float* rm1,
                         const float* rv1, float* h1, int total) {
  int idx = blockIdx.x * blockDim.x + threadIdx.x;
  if (idx < total) {
    int f = idx & (HH - 1), n = idx >> 7;
    float v = sacc[n] * W1[f] + b1[f];
    v = (v - rm1[f]) * rsqrtf(rv1[f] + BNEPS) * g1[f] + be1[f];
    h1[idx] = fmaxf(v, 0.0f);
  }
}

// per-feature scatter for layer 2
__global__ void gcn2_scatter(const int* row, const int* col, const float* lin2,
                             const float* dinv, float* acc2, int E, int N) {
  int idx = blockIdx.x * blockDim.x + threadIdx.x;
  int f = idx & (HH - 1);
  int e = idx >> 7;
  if (e < E) {
    int r = row[e], c = col[e];
    atomicAdd(&acc2[c * HH + f], dinv[r] * dinv[c] * lin2[r * HH + f]);
  } else if (e < E + N) {
    int i = e - E;
    atomicAdd(&acc2[i * HH + f], dinv[i] * dinv[i] * lin2[i * HH + f]);
  }
}

__global__ void bn_relu(const float* acc, const float* b, const float* g,
                        const float* be, const float* rm, const float* rv,
                        float* out, int total) {
  int idx = blockIdx.x * blockDim.x + threadIdx.x;
  if (idx < total) {
    int f = idx & (HH - 1);
    float v = acc[idx] + b[f];
    v = (v - rm[f]) * rsqrtf(rv[f] + BNEPS) * g[f] + be[f];
    out[idx] = fmaxf(v, 0.0f);
  }
}

// ef16[t] = [h2[row], h2[col], attr, 0-pad] as f16, stride KEF
__global__ void build_ef(const float* h2, const int* row, const int* col,
                         const float* attr, _Float16* ef) {
  int t = blockIdx.x;
  int f = threadIdx.x;          // 0..127
  _Float16* e = ef + (size_t)t * KEF;
  int r = row[t], c = col[t];
  e[f]        = (_Float16)h2[r * HH + f];
  e[HH + f]   = (_Float16)h2[c * HH + f];
  if (f < 2)  e[2 * HH + f] = (_Float16)attr[t * 2 + f];
  if (f < KEF - 258) e[258 + f] = (_Float16)0.0f;   // pad 258..287
}

// ---------------------------------------------------- WMMA GEMM machinery
// Pack B (K x N, optionally given as W[N][K] i.e. transposed) into this code's wave32
// B-operand layout (contiguous k, same as A): tile (kt,nt) -> 32 lanes x 16 halfs contiguous.
// lane: n = lane&15, g = lane>>4 ; half h -> K = kt*32 + g*16 + h, col = nt*16+n.
__global__ void pack_b(const float* W, _Float16* Bpk, int Ktiles, int Ksrc,
                       int ldw, int transposed) {
  int lane = threadIdx.x;
  int nt = blockIdx.x, kt = blockIdx.y;
  int n = nt * 16 + (lane & 15);
  int g = lane >> 4;
  _Float16* dst = Bpk + ((size_t)(nt * Ktiles + kt) * 32 + lane) * 16;
#pragma unroll
  for (int h = 0; h < 16; ++h) {
    int k = kt * 32 + g * 16 + h;
    float v = 0.0f;
    if (k < Ksrc)
      v = transposed ? W[(size_t)n * ldw + k] : W[(size_t)k * ldw + n];
    dst[h] = (_Float16)v;
  }
}

// C[M x Nout] = A(f16, row-major, lda) x Bpk (+ bias[Nout]).
// One wave computes a 16(M) x 64(N) strip: the A fragment is loaded once per
// K-step and reused by 4 WMMAs (4x less A traffic than tile-per-wave).
// grid = (Nout/64, M/16), block = 32.
__global__ __launch_bounds__(32)
void wmma_gemm4(const _Float16* __restrict__ A, const _Float16* __restrict__ Bpk,
                const float* __restrict__ bias, float* __restrict__ C,
                int Ktiles, int lda, int ldc) {
  const int lane = threadIdx.x;
  const int n16  = lane & 15;
  const int g    = lane >> 4;
  const int mt   = blockIdx.y;
  const int nt0  = blockIdx.x * 4;           // first of 4 N-tiles

  // A: row m = mt*16 + (lane&15); K-half g: 16 contiguous halfs.
  const _Float16* arow = A + (size_t)(mt * 16 + n16) * lda + g * 16;
  const _Float16* bbase = Bpk + ((size_t)nt0 * Ktiles * 32 + lane) * 16;
  const size_t bstride = (size_t)Ktiles * 512;   // halfs between N-tiles

  v8f acc[4];
#pragma unroll
  for (int nn = 0; nn < 4; ++nn) {
    float bv = bias ? bias[nt0 * 16 + nn * 16 + n16] : 0.0f;
#pragma unroll
    for (int i = 0; i < 8; ++i) acc[nn][i] = bv;
  }

  for (int kt = 0; kt < Ktiles; ++kt) {
    v8h alo = *(const v8h*)(arow + kt * 32);        // K = k0+g*16 .. +7
    v8h ahi = *(const v8h*)(arow + kt * 32 + 8);    // K = k0+g*16+8 .. +15
    v16h a = __builtin_shufflevector(alo, ahi, 0, 1, 2, 3, 4, 5, 6, 7,
                                     8, 9, 10, 11, 12, 13, 14, 15);
#pragma unroll
    for (int nn = 0; nn < 4; ++nn) {
      const _Float16* bk = bbase + (size_t)nn * bstride + (size_t)kt * 512;
      v8h blo = *(const v8h*)(bk);
      v8h bhi = *(const v8h*)(bk + 8);
      v16h b = __builtin_shufflevector(blo, bhi, 0, 1, 2, 3, 4, 5, 6, 7,
                                       8, 9, 10, 11, 12, 13, 14, 15);
      acc[nn] = __builtin_amdgcn_wmma_f32_16x16x32_f16(
          false, a, false, b, (short)0, acc[nn], false, false);
    }
  }

  // C layout per tile: VGPR v -> row M = v + 8*g, col N = lane&15
#pragma unroll
  for (int nn = 0; nn < 4; ++nn) {
    float* crow = C + (size_t)(mt * 16 + 8 * g) * ldc + (nt0 + nn) * 16 + n16;
#pragma unroll
    for (int v = 0; v < 8; ++v) crow[(size_t)v * ldc] = acc[nn][v];
  }
}

// ------------------------------------------------------------- LSTM scan
// One persistent workgroup, 1024 threads (32 waves on a WGP).
// Thread j streams column j of WhhT (coalesced, L2-resident 1MB).
// 4 independent partial accumulators break the FMA dependency chain so the
// dual-issue VALU path can engage.
__global__ __launch_bounds__(1024)
void lstm_scan(const float* __restrict__ pre, const float* __restrict__ whhT,
               float* __restrict__ sout, int T) {
  __shared__ float hsh[LHH];
  __shared__ float gsh[G4];
  const int j = threadIdx.x;
  float c = 0.0f;
  if (j < LHH) hsh[j] = 0.0f;
  __syncthreads();

  for (int t = 0; t < T; ++t) {
    const float* wc = whhT + j;
    float a0 = pre[(size_t)t * G4 + j], a1 = 0.0f, a2 = 0.0f, a3 = 0.0f;
#pragma unroll 8
    for (int k = 0; k < LHH; k += 4) {
      a0 += hsh[k + 0] * wc[(size_t)(k + 0) * G4];
      a1 += hsh[k + 1] * wc[(size_t)(k + 1) * G4];
      a2 += hsh[k + 2] * wc[(size_t)(k + 2) * G4];
      a3 += hsh[k + 3] * wc[(size_t)(k + 3) * G4];
    }
    gsh[j] = (a0 + a1) + (a2 + a3);
    __syncthreads();
    if (j < LHH) {
      float ig = 1.0f / (1.0f + __expf(-gsh[j]));
      float fg = 1.0f / (1.0f + __expf(-gsh[LHH + j]));
      float gg = tanhf(gsh[2 * LHH + j]);
      float og = 1.0f / (1.0f + __expf(-gsh[3 * LHH + j]));
      c = fg * c + ig * gg;
      float h = og * tanhf(c);
      hsh[j] = h;
      sout[(size_t)t * LHH + j] = h;
    }
    __syncthreads();
  }
}

// ------------------------------------------------------------- MLP head
__global__ void head_mlp(const float* __restrict__ s2, const float* __restrict__ lw1,
                         const float* __restrict__ lb1, const float* __restrict__ lw2,
                         const float* __restrict__ lb2, float* __restrict__ out) {
  __shared__ float red[HH];
  const int t = blockIdx.x;
  const int i = threadIdx.x;              // 0..127
  const float* s = s2 + (size_t)t * LHH;
  float a0 = lb1[i], a1 = 0.0f;
#pragma unroll 8
  for (int k = 0; k < LHH; k += 2) {
    a0 += s[k] * lw1[k * HH + i];
    a1 += s[k + 1] * lw1[(k + 1) * HH + i];
  }
  float a = fmaxf(a0 + a1, 0.0f);
  red[i] = a * lw2[i];
  __syncthreads();
  for (int stride = HH / 2; stride > 0; stride >>= 1) {
    if (i < stride) red[i] += red[i + stride];
    __syncthreads();
  }
  if (i == 0) out[t] = red[0] + lb2[0];
}

// ---------------------------------------------------------------- launch
extern "C" void kernel_launch(void* const* d_in, const int* in_sizes, int n_in,
                              void* d_out, int out_size, void* d_ws, size_t ws_size,
                              hipStream_t stream) {
  const float* x     = (const float*)d_in[0];
  const int*   ei    = (const int*)d_in[1];
  const int*   row   = ei;
  const int*   col   = ei + EE;
  const float* eattr = (const float*)d_in[2];
  const float* W1  = (const float*)d_in[3],  *b1  = (const float*)d_in[4];
  const float* g1  = (const float*)d_in[5],  *be1 = (const float*)d_in[6];
  const float* rm1 = (const float*)d_in[7],  *rv1 = (const float*)d_in[8];
  const float* W2  = (const float*)d_in[9],  *b2  = (const float*)d_in[10];
  const float* g2  = (const float*)d_in[11], *be2 = (const float*)d_in[12];
  const float* rm2 = (const float*)d_in[13], *rv2 = (const float*)d_in[14];
  const float* Wih0 = (const float*)d_in[15], *Whh0 = (const float*)d_in[16];
  const float* bih0 = (const float*)d_in[17], *bhh0 = (const float*)d_in[18];
  const float* Wih1 = (const float*)d_in[19], *Whh1 = (const float*)d_in[20];
  const float* bih1 = (const float*)d_in[21], *bhh1 = (const float*)d_in[22];
  const float* lw1 = (const float*)d_in[23], *lb1 = (const float*)d_in[24];
  const float* lw2 = (const float*)d_in[25], *lb2 = (const float*)d_in[26];
  float* out = (float*)d_out;

  // ---- workspace carve-up (256B aligned)
  char* ws = (char*)d_ws;
  size_t off = 0;
  auto alloc = [&](size_t bytes) -> char* {
    char* p = ws + off;
    off += (bytes + 255) & ~(size_t)255;
    return p;
  };
  float*     dinv   = (float*)alloc((size_t)NN * 4);
  float*     sacc   = (float*)alloc((size_t)NN * 4);
  float*     h1     = (float*)alloc((size_t)NN * HH * 4);
  float*     lin2   = (float*)alloc((size_t)NN * HH * 4);
  float*     acc2   = (float*)alloc((size_t)NN * HH * 4);
  float*     h2     = (float*)alloc((size_t)NN * HH * 4);
  _Float16*  a16    = (_Float16*)alloc((size_t)EE * KEF * 2);   // reused h1/ef/s1
  _Float16*  w2pk   = (_Float16*)alloc((size_t)HH * HH * 2);
  _Float16*  wih0pk = (_Float16*)alloc((size_t)KEF * G4 * 2);
  _Float16*  wih1pk = (_Float16*)alloc((size_t)LHH * G4 * 2);
  float*     bsum0  = (float*)alloc((size_t)G4 * 4);
  float*     bsum1  = (float*)alloc((size_t)G4 * 4);
  float*     whh0T  = (float*)alloc((size_t)LHH * G4 * 4);
  float*     whh1T  = (float*)alloc((size_t)LHH * G4 * 4);
  float*     pre    = (float*)alloc((size_t)EE * G4 * 4);       // reused L0/L1
  float*     s1     = (float*)alloc((size_t)EE * LHH * 4);
  float*     s2     = (float*)alloc((size_t)EE * LHH * 4);

  // ---- weight prep (every call; deterministic)
  pack_b<<<dim3(HH / 16, HH / 32), 32, 0, stream>>>(W2, w2pk, 4, 128, 128, 0);
  pack_b<<<dim3(G4 / 16, KEF / 32), 32, 0, stream>>>(Wih0, wih0pk, 9, 258, 258, 1);
  pack_b<<<dim3(G4 / 16, LHH / 32), 32, 0, stream>>>(Wih1, wih1pk, 8, 256, 256, 1);
  add2_f32<<<(G4 + 255) / 256, 256, 0, stream>>>(bih0, bhh0, bsum0, G4);
  add2_f32<<<(G4 + 255) / 256, 256, 0, stream>>>(bih1, bhh1, bsum1, G4);
  transp_whh<<<(LHH * G4 + 255) / 256, 256, 0, stream>>>(Whh0, whh0T);
  transp_whh<<<(LHH * G4 + 255) / 256, 256, 0, stream>>>(Whh1, whh1T);

  // ---- GCN layer 1 (rank-1 collapse)
  zero_f32<<<(NN + 255) / 256, 256, 0, stream>>>(dinv, NN);
  deg_edges<<<(EE + 255) / 256, 256, 0, stream>>>(col, dinv, EE);
  dinv_fin<<<(NN + 255) / 256, 256, 0, stream>>>(dinv, NN);
  zero_f32<<<(NN + 255) / 256, 256, 0, stream>>>(sacc, NN);
  gcn1_scatter<<<(EE + NN + 255) / 256, 256, 0, stream>>>(row, col, x, dinv, sacc, EE, NN);
  h1_fused<<<(NN * HH + 255) / 256, 256, 0, stream>>>(sacc, W1, b1, g1, be1, rm1, rv1,
                                                      h1, NN * HH);

  // ---- GCN layer 2: WMMA GEMM + scatter + BN
  cvt_f32_f16<<<(NN * HH + 255) / 256, 256, 0, stream>>>(h1, a16, NN * HH);
  wmma_gemm4<<<dim3(HH / 64, NN / 16), 32, 0, stream>>>(a16, w2pk, nullptr, lin2,
                                                        4, HH, HH);
  zero_f32<<<(NN * HH + 255) / 256, 256, 0, stream>>>(acc2, NN * HH);
  gcn2_scatter<<<((EE + NN) * HH + 255) / 256, 256, 0, stream>>>(row, col, lin2, dinv,
                                                                 acc2, EE, NN);
  bn_relu<<<(NN * HH + 255) / 256, 256, 0, stream>>>(acc2, b2, g2, be2, rm2, rv2,
                                                     h2, NN * HH);

  // ---- edge features -> f16, LSTM layer 0 input GEMM, serial scan
  build_ef<<<EE, HH, 0, stream>>>(h2, row, col, eattr, a16);
  wmma_gemm4<<<dim3(G4 / 64, EE / 16), 32, 0, stream>>>(a16, wih0pk, bsum0, pre,
                                                        KEF / 32, KEF, G4);
  lstm_scan<<<1, 1024, 0, stream>>>(pre, whh0T, s1, EE);

  // ---- LSTM layer 1
  cvt_f32_f16<<<(EE * LHH + 255) / 256, 256, 0, stream>>>(s1, a16, EE * LHH);
  wmma_gemm4<<<dim3(G4 / 64, EE / 16), 32, 0, stream>>>(a16, wih1pk, bsum1, pre,
                                                        LHH / 32, LHH, G4);
  lstm_scan<<<1, 1024, 0, stream>>>(pre, whh1T, s2, EE);

  // ---- MLP head
  head_mlp<<<EE, HH, 0, stream>>>(s2, lw1, lb1, lw2, lb2, out);
  (void)in_sizes; (void)n_in; (void)out_size; (void)ws_size;
}
